// SRModel_32323923869897
// MI455X (gfx1250) — matrix operand order repaired and checked by reading, not compiled, not run on hardware
//
#include <hip/hip_runtime.h>
#include <hip/hip_bf16.h>

// Problem constants
#define B_ 16
#define T_ 1024
#define F_ 80
#define D_ 512
#define M_ (B_*T_)          // 16384 rows
#define FP_ 96              // F padded to multiple of 32
#define VOUT 29             // V+1
#define MW 4                // M-tiles per wave in GEMM (64 rows per block-Y)

typedef __attribute__((ext_vector_type(16))) __bf16 v16bf;
typedef __attribute__((ext_vector_type(8)))  __bf16 v8bf;
typedef __attribute__((ext_vector_type(8)))  float  v8f;

static __device__ inline __bf16 f2bf(float x) {
    unsigned u = __float_as_uint(x);
    u = (u + 0x7FFFu + ((u >> 16) & 1u)) >> 16;   // round-to-nearest-even
    unsigned short s = (unsigned short)u;
    __bf16 r;
    __builtin_memcpy(&r, &s, 2);
    return r;
}

static __device__ inline float sigf(float x) { return 1.f / (1.f + __expf(-x)); }

// ---------------------------------------------------------------------------
// Weight conversion kernels
// ---------------------------------------------------------------------------
// dst[n*Kp + k] = (n<N && k<K) ? src[k*N + n] : 0      (transpose + pad + bf16)
__global__ void convT_bf16(const float* __restrict__ src, int K, int N,
                           __bf16* __restrict__ dst, int Np, int Kp) {
    int idx = blockIdx.x * blockDim.x + threadIdx.x;
    int tot = Np * Kp;
    if (idx >= tot) return;
    int n = idx / Kp, k = idx % Kp;
    float v = (n < N && k < K) ? src[k * N + n] : 0.f;
    dst[idx] = f2bf(v);
}

// straight f32 -> bf16 copy (weight already [N,K])
__global__ void conv_bf16(const float* __restrict__ src, __bf16* __restrict__ dst, int n) {
    int idx = blockIdx.x * blockDim.x + threadIdx.x;
    if (idx < n) dst[idx] = f2bf(src[idx]);
}

// ---------------------------------------------------------------------------
// LayerNorm -> bf16 (one wave per row).  in_mode 1: read x as (B,T,F) with
// row = t*B + b time-major; in_mode 0: in[row*K + i].
// ---------------------------------------------------------------------------
__global__ void ln_bf16(const float* __restrict__ in,
                        const float* __restrict__ gam,
                        const float* __restrict__ bet,
                        __bf16* __restrict__ out,
                        int K, int Kp, int in_mode) {
    const int lane = threadIdx.x & 31;
    const int row  = blockIdx.x * 8 + (threadIdx.x >> 5);
    const int nIter = Kp >> 5;                 // Kp multiple of 32
    float vals[16];
    float s = 0.f, sq = 0.f;
#pragma unroll
    for (int it = 0; it < 16; ++it) {
        if (it >= nIter) break;
        int i = it * 32 + lane;
        float v = 0.f;
        if (i < K) {
            if (in_mode == 1) { int tt = row >> 4, bb = row & 15; v = in[(bb * T_ + tt) * K + i]; }
            else              { v = in[row * K + i]; }
        }
        vals[it] = v; s += v; sq += v * v;
    }
#pragma unroll
    for (int off = 16; off > 0; off >>= 1) {
        s  += __shfl_xor(s,  off, 32);
        sq += __shfl_xor(sq, off, 32);
    }
    float mean = s / (float)K;
    float var  = sq / (float)K - mean * mean;
    float rstd = rsqrtf(var + 1e-5f);
#pragma unroll
    for (int it = 0; it < 16; ++it) {
        if (it >= nIter) break;
        int i = it * 32 + lane;
        float o = (i < K) ? ((vals[it] - mean) * rstd * gam[i] + bet[i]) : 0.f;
        out[row * Kp + i] = f2bf(o);
    }
}

// ---------------------------------------------------------------------------
// bf16 WMMA GEMM:  C[M,N] = A[M,K] * Bt^T    (Bt stored [N,K] row-major bf16)
// One wave computes a (MW*16)x16 column of tiles, reusing each B fragment for
// MW WMMAs (4x less weight traffic from L2). 8 waves/block span 128 columns.
// out_mode 0: C[gm*ldc + gn] (f32 row-major)
// out_mode 1: classifier — row gm = t*B + b, write out[(b*T + t)*29 + gn], gn<29
// ---------------------------------------------------------------------------
__global__ void gemm_bf16_wmma(const __bf16* __restrict__ A, int lda,
                               const __bf16* __restrict__ Bt, int ldb,
                               float* __restrict__ C, int ldc,
                               int N, int K,
                               const float* __restrict__ bias0,
                               const float* __restrict__ bias1,
                               int out_mode) {
    const int lane  = threadIdx.x & 31;
    const int wave  = threadIdx.x >> 5;
    const int ntile = blockIdx.x * 8 + wave;
    const int mbase = blockIdx.y * (16 * MW);
    if (ntile * 16 >= N) return;

    const int l16  = lane & 15;
    const int half = lane >> 4;
    const __bf16* arow = A  + (mbase + l16) * lda;        // A: lane holds row M=l16
    const __bf16* brow = Bt + (ntile * 16 + l16) * ldb;   // B: lane holds col N=l16

    v8f acc[MW] = {{}, {}, {}, {}};
    for (int kk = 0; kk < K; kk += 32) {
        v16bf bfr = *(const v16bf*)(brow + kk + half * 16);    // K = kk + 16h .. +16
#pragma unroll
        for (int m = 0; m < MW; ++m) {
            union { v16bf v; v8bf h[2]; } a;
            const __bf16* ap = arow + m * 16 * lda + kk + half * 16;
            a.h[0] = *(const v8bf*)(ap);                       // K = kk + 16h .. +8
            a.h[1] = *(const v8bf*)(ap + 8);                   // K = kk+16h+8 .. +8
            acc[m] = __builtin_amdgcn_wmma_f32_16x16x32_bf16(
                         false, a.v, false, bfr, (short)0, acc[m], false, false);
        }
    }

    const int gn = ntile * 16 + l16;
    float bsum = 0.f;
    if (bias0) bsum += bias0[gn];
    if (bias1) bsum += bias1[gn];
#pragma unroll
    for (int m = 0; m < MW; ++m) {
#pragma unroll
        for (int r = 0; r < 8; ++r) {
            int gm = mbase + m * 16 + r + 8 * half;            // C/D layout
            float v = acc[m][r] + bsum;
            if (out_mode == 0) {
                C[gm * ldc + gn] = v;
            } else {
                if (gn < VOUT) {
                    int bb = gm & 15, tt = gm >> 4;
                    C[(bb * T_ + tt) * VOUT + gn] = v;
                }
            }
        }
    }
}

// ---------------------------------------------------------------------------
// SRU elementwise recurrence: one thread per (b,d); sequential over T.
// hres_in == nullptr -> xres lives in U[..., 3D:4D] (layer 0).
// ---------------------------------------------------------------------------
__global__ void sru_scan(const float* __restrict__ U, int ldu,
                         const float* __restrict__ hres_in,
                         const float* __restrict__ wc,
                         const float* __restrict__ bias,
                         float* __restrict__ hout) {
    int tid = blockIdx.x * blockDim.x + threadIdx.x;
    if (tid >= B_ * D_) return;
    int b = tid / D_, d = tid % D_;
    float vf = wc[d], vr = wc[D_ + d];
    float bf = bias[d], br = bias[D_ + d];
    float c = 0.f;
    for (int t = 0; t < T_; ++t) {
        int base = (t * B_ + b) * ldu;
        float u0 = U[base + d];
        float u1 = U[base + D_ + d];
        float u2 = U[base + 2 * D_ + d];
        float xr = hres_in ? hres_in[(t * B_ + b) * D_ + d] : U[base + 3 * D_ + d];
        float f = sigf(u1 + vf * c + bf);
        float r = sigf(u2 + vr * c + br);
        c = f * c + (1.f - f) * u0;
        hout[(t * B_ + b) * D_ + d] = r * c + (1.f - r) * xr;
    }
}

// ---------------------------------------------------------------------------
// LSTM recurrence: ONE workgroup of 1024 threads = 32 waves on a single WGP.
// Wave w owns output columns [w*16, w*16+16) of D for all 4 gates.
// h state lives in LDS as bf16 in WMMA A-fragment-friendly layout (16 KB);
// c state lives entirely in registers (each wave owns its columns).
// Hardware s_barrier (__syncthreads) separates read-h / write-h phases —
// no global atomics, no global state traffic in the sequential loop.
// ---------------------------------------------------------------------------
__global__ __launch_bounds__(1024, 1)
void lstm_rec(const float* __restrict__ gx,        // [T*B, 4D] biases folded in
              const __bf16* __restrict__ whh,      // [4D, D] bf16
              float* __restrict__ hout) {          // [T*B, D]
    __shared__ __bf16 hbf[B_ * D_];                // 16 KB, row-major [b][k]

    const int lane  = threadIdx.x & 31;
    const int wave  = threadIdx.x >> 5;            // 0..31 == dtile
    const int l16   = lane & 15;
    const int half  = lane >> 4;
    const int dtile = wave;

    // zero initial h state
    for (int i = threadIdx.x; i < B_ * D_; i += 1024) hbf[i] = f2bf(0.f);
    float creg[8];
#pragma unroll
    for (int r = 0; r < 8; ++r) creg[r] = 0.f;
    __syncthreads();

    const __bf16* arow = hbf + l16 * D_;           // A row = batch l16
    const int gn = dtile * 16 + l16;               // column in D

    for (int t = 0; t < T_; ++t) {
        v8f acc[4] = {{}, {}, {}, {}};
        for (int kk = 0; kk < D_; kk += 32) {
            union { v16bf v; v8bf h[2]; } a;       // A fragment from LDS
            a.h[0] = *(const v8bf*)(arow + kk + half * 16);
            a.h[1] = *(const v8bf*)(arow + kk + half * 16 + 8);
#pragma unroll
            for (int g4 = 0; g4 < 4; ++g4) {
                int n = g4 * D_ + dtile * 16 + l16;
                v16bf bfr = *(const v16bf*)(whh + n * D_ + kk + half * 16);
                acc[g4] = __builtin_amdgcn_wmma_f32_16x16x32_bf16(
                              false, a.v, false, bfr, (short)0, acc[g4], false, false);
            }
        }

        float h2v[8];
#pragma unroll
        for (int r = 0; r < 8; ++r) {
            int gm = r + 8 * half;                 // batch row
            int gbase = (t * B_ + gm) * (4 * D_) + gn;
            float gi = acc[0][r] + gx[gbase];
            float gf = acc[1][r] + gx[gbase + D_];
            float gg = acc[2][r] + gx[gbase + 2 * D_];
            float go = acc[3][r] + gx[gbase + 3 * D_];
            float c2 = sigf(gf) * creg[r] + sigf(gi) * tanhf(gg);
            creg[r] = c2;
            h2v[r] = sigf(go) * tanhf(c2);
        }

        __syncthreads();                           // all h reads of step t done
#pragma unroll
        for (int r = 0; r < 8; ++r) {
            int gm = r + 8 * half;
            hbf[gm * D_ + gn] = f2bf(h2v[r]);
            hout[(t * B_ + gm) * D_ + gn] = h2v[r];
        }
        __syncthreads();                           // h writes visible
    }
}

// ---------------------------------------------------------------------------
// Host launcher
// ---------------------------------------------------------------------------
extern "C" void kernel_launch(void* const* d_in, const int* in_sizes, int n_in,
                              void* d_out, int out_size, void* d_ws, size_t ws_size,
                              hipStream_t stream) {
    (void)in_sizes; (void)n_in; (void)out_size; (void)ws_size;
    const float* x        = (const float*)d_in[0];
    const float* sru_w0   = (const float*)d_in[1];
    const float* sru_wc0  = (const float*)d_in[2];
    const float* sru_b0   = (const float*)d_in[3];
    const float* sru_ln0g = (const float*)d_in[4];
    const float* sru_ln0b = (const float*)d_in[5];
    const float* sru_w    = (const float*)d_in[6];   // (7, D, 3D)
    const float* sru_wc   = (const float*)d_in[7];   // (7, 2D)
    const float* sru_b    = (const float*)d_in[8];
    const float* sru_lng  = (const float*)d_in[9];   // (7, D)
    const float* sru_lnb  = (const float*)d_in[10];
    const float* lstm_lng = (const float*)d_in[11];  // (3, D)
    const float* lstm_lnb = (const float*)d_in[12];
    const float* lstm_wih = (const float*)d_in[13];  // (3, 4D, D)
    const float* lstm_whh = (const float*)d_in[14];  // (3, 4D, D)
    const float* lstm_bih = (const float*)d_in[15];  // (3, 4D)
    const float* lstm_bhh = (const float*)d_in[16];
    const float* cls_lng  = (const float*)d_in[17];
    const float* cls_lnb  = (const float*)d_in[18];
    const float* cls_w    = (const float*)d_in[19];  // (D, 29)
    float* out = (float*)d_out;

    // ---- workspace layout (256B aligned) ----
    char* ws = (char*)d_ws;
    size_t off = 0;
    auto alloc = [&](size_t bytes) { void* p = ws + off; off = (off + bytes + 255) & ~(size_t)255; return p; };
    __bf16* Wt0   = (__bf16*)alloc((size_t)2048 * FP_ * 2);          // [2048, 96]
    __bf16* Wt    = (__bf16*)alloc((size_t)7 * 1536 * D_ * 2);       // 7 x [1536, 512]
    __bf16* WihB  = (__bf16*)alloc((size_t)3 * 2048 * D_ * 2);       // 3 x [2048, 512]
    __bf16* WhhB  = (__bf16*)alloc((size_t)3 * 2048 * D_ * 2);
    __bf16* WclsT = (__bf16*)alloc((size_t)32 * D_ * 2);             // [32, 512]
    __bf16* xn    = (__bf16*)alloc((size_t)M_ * D_ * 2);             // bf16 LN activations
    float*  U     = (float*)alloc((size_t)M_ * 2048 * 4);            // U / gx buffer
    float*  hA    = (float*)alloc((size_t)M_ * D_ * 4);
    float*  hB    = (float*)alloc((size_t)M_ * D_ * 4);

    const dim3 blk256(256);
    auto gs = [](int n) { return dim3((n + 255) / 256); };

    // ---- weight conversion ----
    convT_bf16<<<gs(2048 * FP_), blk256, 0, stream>>>(sru_w0, F_, 4 * D_, Wt0, 2048, FP_);
    for (int i = 0; i < 7; ++i)
        convT_bf16<<<gs(1536 * D_), blk256, 0, stream>>>(
            sru_w + (size_t)i * D_ * 1536, D_, 1536, Wt + (size_t)i * 1536 * D_, 1536, D_);
    for (int l = 0; l < 3; ++l) {
        conv_bf16<<<gs(2048 * D_), blk256, 0, stream>>>(
            lstm_wih + (size_t)l * 2048 * D_, WihB + (size_t)l * 2048 * D_, 2048 * D_);
        conv_bf16<<<gs(2048 * D_), blk256, 0, stream>>>(
            lstm_whh + (size_t)l * 2048 * D_, WhhB + (size_t)l * 2048 * D_, 2048 * D_);
    }
    convT_bf16<<<gs(32 * D_), blk256, 0, stream>>>(cls_w, D_, VOUT, WclsT, 32, D_);

    const dim3 lnGrid(M_ / 8);

    // ---- SRU layer 0: LN(F) -> GEMM(96 -> 2048) -> scan ----
    ln_bf16<<<lnGrid, blk256, 0, stream>>>(x, sru_ln0g, sru_ln0b, xn, F_, FP_, 1);
    gemm_bf16_wmma<<<dim3(16, M_ / (16 * MW)), blk256, 0, stream>>>(
        xn, FP_, Wt0, FP_, U, 2048, 2048, FP_, nullptr, nullptr, 0);
    sru_scan<<<gs(B_ * D_), blk256, 0, stream>>>(U, 2048, nullptr, sru_wc0, sru_b0, hA);

    // ---- SRU layers 1..7 ----
    float* hcur = hA;
    float* hnext = hB;
    for (int i = 0; i < 7; ++i) {
        ln_bf16<<<lnGrid, blk256, 0, stream>>>(
            hcur, sru_lng + i * D_, sru_lnb + i * D_, xn, D_, D_, 0);
        gemm_bf16_wmma<<<dim3(12, M_ / (16 * MW)), blk256, 0, stream>>>(
            xn, D_, Wt + (size_t)i * 1536 * D_, D_, U, 1536, 1536, D_, nullptr, nullptr, 0);
        sru_scan<<<gs(B_ * D_), blk256, 0, stream>>>(
            U, 1536, hcur, sru_wc + i * 2 * D_, sru_b + i * 2 * D_, hnext);
        float* tmp = hcur; hcur = hnext; hnext = tmp;
    }

    // ---- LSTM layers ----
    for (int l = 0; l < 3; ++l) {
        ln_bf16<<<lnGrid, blk256, 0, stream>>>(
            hcur, lstm_lng + l * D_, lstm_lnb + l * D_, xn, D_, D_, 0);
        gemm_bf16_wmma<<<dim3(16, M_ / (16 * MW)), blk256, 0, stream>>>(
            xn, D_, WihB + (size_t)l * 2048 * D_, D_, U, 2048, 2048, D_,
            lstm_bih + l * 2048, lstm_bhh + l * 2048, 0);
        lstm_rec<<<dim3(1), dim3(1024), 0, stream>>>(
            U, WhhB + (size_t)l * 2048 * D_, hnext);
        float* tmp = hcur; hcur = hnext; hnext = tmp;
    }

    // ---- classifier: LN -> GEMM(512 -> 29), permuted (b,t) output ----
    ln_bf16<<<lnGrid, blk256, 0, stream>>>(hcur, cls_lng, cls_lnb, xn, D_, D_, 0);
    gemm_bf16_wmma<<<dim3(1, M_ / (16 * MW)), blk256, 0, stream>>>(
        xn, D_, WclsT, D_, out, VOUT, 32, D_, nullptr, nullptr, 1);
}
